// AdaptiveESN_62113817035102
// MI455X (gfx1250) — matrix operand order repaired and checked by reading, not compiled, not run on hardware
//
#include <hip/hip_runtime.h>

// ---- problem constants ----
#define BB 64
#define TT 2048
#define DD 128
#define HH 512
#define LEAKY 0.26f

// ---- launch shape ----
#define NWG   8      // workgroups; each owns HH/NWG = 64 state columns
#define HS    64     // H slice per WG
#define DSL   16     // D slice per WG for readout (8*16 = 128)
#define NTHR  512    // 16 wave32 per WG
#define NWAVE 16

// ---- LDS layout (bf16, padded strides keep 16B alignment, dodge bank conflicts) ----
#define SW 520       // stride (elements) for K=512 arrays
#define SX 136       // stride (elements) for K=128 arrays
#define OFF_WRES 0
#define OFF_WIN  (OFF_WRES + HS * SW * 2)   // 66560
#define OFF_WRO  (OFF_WIN  + HS * SX * 2)   // +17408
#define OFF_HBF  (OFF_WRO  + DSL * SW * 2)  // +16640
#define OFF_XBF  (OFF_HBF  + BB * SW * 2)   // +66560
#define OFF_BIAS (OFF_XBF  + BB * SX * 2)   // +17408
#define LDS_BYTES (OFF_BIAS + HS * 4)       // 184832 bytes < 320KB/WGP

typedef __attribute__((ext_vector_type(16))) __bf16 v16bf;
typedef __attribute__((ext_vector_type(8)))  __bf16 v8bf;
typedef __attribute__((ext_vector_type(4)))  __bf16 v4bf;
typedef __attribute__((ext_vector_type(8)))  float  v8f;
typedef __attribute__((ext_vector_type(4)))  float  v4f;

#if __has_builtin(__builtin_amdgcn_tanhf)
#define TANHF(v) __builtin_amdgcn_tanhf(v)    // native v_tanh_f32 on gfx1250
#else
#define TANHF(v) tanhf(v)
#endif

// A fragment: 16x32 bf16, row-major LDS [M][K(stride)], per ISA layout:
// lane(l<16): M=l, K = k0 + {0..7, 16..23}; lane(l>=16): M=l-16, K = k0 + {8..15, 24..31}
static __device__ __forceinline__ v16bf frag_a(const __bf16* A, int stride, int mBase,
                                               int k0, int row, int kh) {
  const __bf16* p = A + (mBase + row) * stride + k0 + kh * 8;
  v8bf lo = *(const v8bf*)p;
  v8bf hi = *(const v8bf*)(p + 16);
  return __builtin_shufflevector(lo, hi, 0,1,2,3,4,5,6,7,8,9,10,11,12,13,14,15);
}

// B fragment: 32x16 bf16 with weights stored [N][K(stride)] in LDS:
// lane(l<16): N=l, K = k0 + {0..7, 16..23}; lane(l>=16): N=l-16, K = k0 + {8..15, 24..31}
static __device__ __forceinline__ v16bf frag_b(const __bf16* W, int stride, int nBase,
                                               int k0, int row, int kh) {
  const __bf16* p = W + (nBase + row) * stride + k0 + kh * 8;
  v8bf lo = *(const v8bf*)p;
  v8bf hi = *(const v8bf*)(p + 16);
  return __builtin_shufflevector(lo, hi, 0,1,2,3,4,5,6,7,8,9,10,11,12,13,14,15);
}

static __device__ __forceinline__ void grid_barrier(unsigned* cnt, unsigned target) {
  __threadfence();          // release: make this WG's h stores visible at device scope
  __syncthreads();          // all threads in WG have fenced
  if (threadIdx.x == 0) {
    __hip_atomic_fetch_add(cnt, 1u, __ATOMIC_RELEASE, __HIP_MEMORY_SCOPE_AGENT);
    while (__hip_atomic_load(cnt, __ATOMIC_ACQUIRE, __HIP_MEMORY_SCOPE_AGENT) < target) {
      __builtin_amdgcn_s_sleep(2);
    }
  }
  __syncthreads();
  __threadfence();          // acquire: invalidate near caches before re-reading h
}

extern "C" __global__ void __launch_bounds__(NTHR, 1)
esn_fused(const float* __restrict__ x, const float* __restrict__ W_in,
          const float* __restrict__ b_in, const float* __restrict__ W_res,
          const float* __restrict__ b_res, const float* __restrict__ W_ro,
          float* __restrict__ y, unsigned char* __restrict__ ws) {
  extern __shared__ char smem[];
  __bf16* sWres = (__bf16*)(smem + OFF_WRES);   // [HS][SW]   rows = output neuron j
  __bf16* sWin  = (__bf16*)(smem + OFF_WIN);    // [HS][SX]
  __bf16* sWro  = (__bf16*)(smem + OFF_WRO);    // [DSL][SW]  rows = output dim d
  __bf16* sH    = (__bf16*)(smem + OFF_HBF);    // [BB][SW]   full h (bf16)
  __bf16* sX    = (__bf16*)(smem + OFF_XBF);    // [BB][SX]   x_t (bf16)
  float*  sBias = (float*)(smem + OFF_BIAS);    // [HS] b_in+b_res slice

  const int g    = blockIdx.x;        // 0..7: owns h cols [g*64, g*64+64), y dims [g*16, g*16+16)
  const int tid  = threadIdx.x;
  const int lane = tid & 31;
  const int wave = tid >> 5;

  unsigned* cnt  = (unsigned*)ws;
  __bf16* hbufA  = (__bf16*)(ws + 256);                       // [BB][HH] bf16 exchange, ping
  __bf16* hbufB  = (__bf16*)(ws + 256 + (size_t)BB * HH * 2); // pong

  // ---- one-time weight preload (f32 -> bf16 into LDS) ----
  for (int i = tid; i < HS * HH; i += NTHR) {
    int r = i >> 9, k = i & (HH - 1);
    sWres[r * SW + k] = (__bf16)W_res[(size_t)(g * HS + r) * HH + k];
  }
  for (int i = tid; i < HS * DD; i += NTHR) {
    int r = i >> 7, k = i & (DD - 1);
    sWin[r * SX + k] = (__bf16)W_in[(size_t)(g * HS + r) * DD + k];
  }
  for (int i = tid; i < DSL * HH; i += NTHR) {
    int r = i >> 9, k = i & (HH - 1);
    sWro[r * SW + k] = (__bf16)W_ro[(size_t)(g * DSL + r) * HH + k];
  }
  if (tid < HS) sBias[tid] = b_in[g * HS + tid] + b_res[g * HS + tid];
  for (int i = tid; i < BB * SW; i += NTHR) sH[i] = (__bf16)0.0f;  // h0 = 0
  __syncthreads();

  // wave -> 16x16 C tile of this WG's [64 x 64] h slice
  const int mi = wave >> 2, ni = wave & 3;
  const int mBase = mi * 16, nBase = ni * 16;
  const int row = lane & 15, kh = lane >> 4;
  const float biasc  = sBias[nBase + row];   // bias depends on column n only
  const float aleak  = LEAKY, naleak = 1.0f - LEAKY;

  v8f hprev = {};  // this wave's h tile (f32) persists in registers for all T steps

  #pragma unroll 1
  for (int t = 0; t < TT; ++t) {
    __bf16* hb = (t & 1) ? hbufB : hbufA;

    // stage x_t -> LDS bf16: float4 NT loads + packed bf16 converts + b64 DS stores
    #pragma unroll
    for (int c = 0; c < (BB * DD / 4) / NTHR; ++c) {       // 4 chunks / thread
      int idx = tid + c * NTHR;                            // 0..2047
      int b = idx >> 5, d4 = (idx & 31) * 4;
      v4f xv = __builtin_nontemporal_load(
          (const v4f*)&x[(size_t)b * TT * DD + (size_t)t * DD + d4]);
      *(v4bf*)&sX[b * SX + d4] = __builtin_convertvector(xv, v4bf);
    }
    // prefetch next timestep's x rows into cache while we compute
    if (t + 1 < TT && tid < BB) {
      const float* px = &x[(size_t)tid * TT * DD + (size_t)(t + 1) * DD];
      __builtin_prefetch(px, 0, 1);
      __builtin_prefetch(px + 64, 0, 1);
    }
    __syncthreads();

    // pre = bias + x_t @ W_in_sliceT + h @ W_res_sliceT
    v8f acc;
    #pragma unroll
    for (int v = 0; v < 8; ++v) acc[v] = biasc;

    #pragma unroll
    for (int k0 = 0; k0 < DD; k0 += 32) {
      v16bf a = frag_a(sX, SX, mBase, k0, row, kh);
      v16bf b = frag_b(sWin, SX, nBase, k0, row, kh);
      acc = __builtin_amdgcn_wmma_f32_16x16x32_bf16(false, a, false, b, (short)0, acc, false, false);
    }
    #pragma unroll 4
    for (int k0 = 0; k0 < HH; k0 += 32) {
      v16bf a = frag_a(sH, SW, mBase, k0, row, kh);
      v16bf b = frag_b(sWres, SW, nBase, k0, row, kh);
      acc = __builtin_amdgcn_wmma_f32_16x16x32_bf16(false, a, false, b, (short)0, acc, false, false);
    }

    // leaky tanh update in registers (native v_tanh_f32); publish bf16 slice to L2
    v8f hnew;
    #pragma unroll
    for (int v = 0; v < 8; ++v) hnew[v] = naleak * hprev[v] + aleak * TANHF(acc[v]);
    hprev = hnew;

    {
      const int col = g * HS + nBase + row;
      #pragma unroll
      for (int v = 0; v < 8; ++v) {
        int m = mBase + v + 8 * kh;                 // C-tile row per ISA layout
        hb[(size_t)m * HH + col] = (__bf16)hnew[v];
      }
    }

    grid_barrier(cnt, (unsigned)gridDim.x * (unsigned)(t + 1));

    // pull full h[t] (bf16, 64KB) from L2 into LDS: 4096 x 16B chunks
    {
      const v8bf* src = (const v8bf*)hb;
      v8bf tmp[8];
      #pragma unroll
      for (int c = 0; c < 8; ++c) {
        int idx = tid + c * NTHR;                   // 0..4095
        tmp[c] = src[(idx >> 6) * (HH / 8) + (idx & 63)];
      }
      #pragma unroll
      for (int c = 0; c < 8; ++c) {
        int idx = tid + c * NTHR;
        *(v8bf*)&sH[(idx >> 6) * SW + (idx & 63) * 8] = tmp[c];
      }
    }
    __syncthreads();

    // fused readout: y[:, t, g*16 .. g*16+16) = h[t] @ W_ro_sliceT  (waves 0..3)
    if (wave < 4) {
      v8f yacc = {};
      #pragma unroll 4
      for (int k0 = 0; k0 < HH; k0 += 32) {
        v16bf a = frag_a(sH, SW, wave * 16, k0, row, kh);
        v16bf b = frag_b(sWro, SW, 0, k0, row, kh);
        yacc = __builtin_amdgcn_wmma_f32_16x16x32_bf16(false, a, false, b, (short)0, yacc, false, false);
      }
      const int dcol = g * DSL + row;
      #pragma unroll
      for (int v = 0; v < 8; ++v) {
        int m = wave * 16 + v + 8 * kh;
        __builtin_nontemporal_store(yacc[v],
            &y[(size_t)m * TT * DD + (size_t)t * DD + dcol]);
      }
    }
    // next iteration only rewrites sX (guarded by its own __syncthreads) and
    // rewrites sH only after the next grid_barrier's __syncthreads.
  }
}

extern "C" void kernel_launch(void* const* d_in, const int* in_sizes, int n_in,
                              void* d_out, int out_size, void* d_ws, size_t ws_size,
                              hipStream_t stream) {
  const float* x     = (const float*)d_in[0];
  const float* W_in  = (const float*)d_in[1];
  const float* b_in  = (const float*)d_in[2];
  const float* W_res = (const float*)d_in[3];
  const float* b_res = (const float*)d_in[4];
  const float* W_ro  = (const float*)d_in[5];
  float* y = (float*)d_out;

  hipFuncSetAttribute(reinterpret_cast<const void*>(&esn_fused),
                      hipFuncAttributeMaxDynamicSharedMemorySize, LDS_BYTES);
  // reset the grid-barrier counter every call (graph-capturable, deterministic)
  hipMemsetAsync(d_ws, 0, 256, stream);
  esn_fused<<<dim3(NWG), dim3(NTHR), LDS_BYTES, stream>>>(
      x, W_in, b_in, W_res, b_res, W_ro, y, (unsigned char*)d_ws);
}
